// PersistentTokenRoutedMLP_76209899700399
// MI455X (gfx1250) — matrix operand order repaired and checked by reading, not compiled, not run on hardware
//
#include <hip/hip_runtime.h>

// ---------------------------------------------------------------------------
// Token-routed MLP (MoE) for MI455X / gfx1250.
//   T=16384 tokens, H=2048, I_e=1024, 8 experts (vocab-bucket routing).
//   ~206 GFLOP vs ~470MB min HBM traffic -> compute bound -> bf16 WMMA path
//   (v_wmma_f32_16x16x32_bf16, fp32 accumulate), fp32->bf16 on the fly.
//   Weight tiles double-buffered in LDS: one barrier per K-step, global
//   loads for tile k+1 overlap the WMMAs for tile k.
// ---------------------------------------------------------------------------

#define NUM_EXPERTS 8
#define HID 2048
#define IE  1024
#define VOCAB 100000
#define TPE (VOCAB / NUM_EXPERTS)   // 12500 tokens per vocab bucket
#define T_TOK 16384
#define M_TILE 32                   // tokens per workgroup
#define N_TILE 64                   // output columns per iteration (4 waves)
#define K_TILE 32                   // K per WMMA step
#define KPAD  (K_TILE + 8)          // LDS pad to break bank conflicts
#define THREADS 256                 // 8 waves (wave32)

typedef __attribute__((ext_vector_type(16))) __bf16         bf16x16;
typedef __attribute__((ext_vector_type(8)))  float          floatx8;
typedef __attribute__((ext_vector_type(8)))  unsigned short us8_t;
typedef __attribute__((ext_vector_type(4)))  unsigned short us4_t;

union BF16x16 { struct { us8_t lo, hi; } u; bf16x16 v; };

__device__ __forceinline__ unsigned short f2bf(float f) {
  unsigned u = __builtin_bit_cast(unsigned, f);
  u += 0x7FFFu + ((u >> 16) & 1u);            // round-to-nearest-even
  return (unsigned short)(u >> 16);
}

// ---------------------------------------------------------------------------
// Routing: bucket token indices per expert. Order within a bucket is
// nondeterministic but each token's math is order-independent -> output
// is deterministic.
// ---------------------------------------------------------------------------
__global__ void zero_counts_kernel(int* p) {
  if (threadIdx.x < NUM_EXPERTS) p[threadIdx.x] = 0;
}

__global__ void route_kernel(const int* __restrict__ tok_ids,
                             int* __restrict__ cnt,
                             int* __restrict__ bucket) {
  int i = blockIdx.x * blockDim.x + threadIdx.x;
  if (i < T_TOK) {
    int id = tok_ids[i];
    id = id < 0 ? 0 : (id >= VOCAB ? VOCAB - 1 : id);
    int e = id / TPE;
    if (e > NUM_EXPERTS - 1) e = NUM_EXPERTS - 1;
    int pos = atomicAdd(&cnt[e], 1);
    bucket[e * T_TOK + pos] = i;
  }
}

// ---------------------------------------------------------------------------
// Fused per-expert MLP: one workgroup = 32 tokens of one expert.
// 8 waves arranged as 2 (M) x 4 (N) grid of 16x16 WMMA accumulators.
// Activation tile (32x2048 bf16, 132KB) + intermediate (32x1024 bf16, 66KB)
// + double-buffered weight tiles (20KB) live in LDS (~219KB of 320KB/WGP).
// ---------------------------------------------------------------------------
__global__ __launch_bounds__(THREADS, 1)
void moe_mlp_kernel(const float* __restrict__ x,
                    const float* __restrict__ gate,
                    const float* __restrict__ up,
                    const float* __restrict__ down,
                    const int* __restrict__ cnt,
                    const int* __restrict__ bucket,
                    float* __restrict__ out) {
  __shared__ unsigned short sX[M_TILE][HID + 16];       // gathered X, bf16
  __shared__ unsigned short sH[M_TILE][IE + 16];        // silu(g)*u, bf16
  __shared__ unsigned short sB[2][2][N_TILE][KPAD];     // [buf][mat][n][k]
  __shared__ int sIdx[M_TILE];

  const int e     = blockIdx.y;
  const int n_e   = cnt[e];
  const int tbase = blockIdx.x * M_TILE;
  if (tbase >= n_e) return;                             // block-uniform exit

  const int tid   = threadIdx.x;
  const int lane  = tid & 31;
  const int wave  = tid >> 5;
  const int m_sub = wave >> 2;                          // 0..1
  const int n_sub = wave & 3;                           // 0..3
  const int l15   = lane & 15;
  const int khalf = (lane & 16) ? 8 : 0;                // A-operand K base

  if (tid < M_TILE) {
    int r = tbase + tid;
    sIdx[tid] = (r < n_e) ? bucket[e * T_TOK + r] : -1;
  }
  __syncthreads();

  // ---- gather + convert activation tile into LDS (once per WG) ----
  {
    int row = tid >> 3;                   // 0..31
    int cb  = (tid & 7) * (HID / 8);      // 256-col span per thread
    int tok = sIdx[row];
    const float* xr = x + (size_t)(tok < 0 ? 0 : tok) * HID + cb;
    for (int c = 0; c < HID / 8; c += 4) {
      float4 f = make_float4(0.f, 0.f, 0.f, 0.f);
      if (tok >= 0) f = *(const float4*)(xr + c);
      us4_t h = { f2bf(f.x), f2bf(f.y), f2bf(f.z), f2bf(f.w) };
      *(us4_t*)&sX[row][cb + c] = h;
    }
  }
  __syncthreads();

  const float* Wg = gate + (size_t)e * HID * IE;
  const float* Wu = up   + (size_t)e * HID * IE;
  const float* Wd = down + (size_t)e * IE * HID;
  const int bn  = tid & 63;                             // staging: column
  const int bkg = tid >> 6;                             // staging: k-group

  // Stage a [K_TILE x N_TILE] fp32 weight tile -> LDS transposed [n][k] bf16.
  // Per thread: 8 column-strided b32 loads, one b128 LDS store.
  auto stageGU = [&](int buf, int kb, int nb) {
    const float* g0 = Wg + (size_t)(kb + bkg * 8) * IE + (nb + bn);
    const float* u0 = Wu + (size_t)(kb + bkg * 8) * IE + (nb + bn);
    us8_t hg, hu;
    #pragma unroll
    for (int i = 0; i < 8; ++i) {
      hg[i] = f2bf(g0[(size_t)i * IE]);
      hu[i] = f2bf(u0[(size_t)i * IE]);
    }
    *(us8_t*)&sB[buf][0][bn][bkg * 8] = hg;
    *(us8_t*)&sB[buf][1][bn][bkg * 8] = hu;
  };
  auto stageD = [&](int buf, int kb, int nb) {
    const float* d0 = Wd + (size_t)(kb + bkg * 8) * HID + (nb + bn);
    us8_t hd;
    #pragma unroll
    for (int i = 0; i < 8; ++i) hd[i] = f2bf(d0[(size_t)i * HID]);
    *(us8_t*)&sB[buf][0][bn][bkg * 8] = hd;
  };

  // ================= Phase A: g = X*Wg, u = X*Wu, H = silu(g)*u ============
  for (int nb = 0; nb < IE; nb += N_TILE) {
    floatx8 Cg = {0.f, 0.f, 0.f, 0.f, 0.f, 0.f, 0.f, 0.f};
    floatx8 Cu = Cg;
    stageGU(0, 0, nb);
    __syncthreads();
    int buf = 0;
    for (int kb = 0; kb < HID; kb += K_TILE, buf ^= 1) {
      if (kb + K_TILE < HID) stageGU(buf ^ 1, kb + K_TILE, nb);  // prefetch

      BF16x16 A, Bg, Bu;
      // A: lane=M row; lanes 0-15 K {0..7,16..23}, lanes 16-31 K {8..15,24..31}
      const unsigned short* ap = &sX[m_sub * 16 + l15][kb + khalf];
      A.u.lo = *(const us8_t*)(ap);
      A.u.hi = *(const us8_t*)(ap + 16);
      // B: lane=N col; lanes 0-15 K {0..7,16..23}, lanes 16-31 K {8..15,24..31}
      const unsigned short* bgp = &sB[buf][0][n_sub * 16 + l15][khalf];
      Bg.u.lo = *(const us8_t*)(bgp);
      Bg.u.hi = *(const us8_t*)(bgp + 16);
      const unsigned short* bup = &sB[buf][1][n_sub * 16 + l15][khalf];
      Bu.u.lo = *(const us8_t*)(bup);
      Bu.u.hi = *(const us8_t*)(bup + 16);

      Cg = __builtin_amdgcn_wmma_f32_16x16x32_bf16(false, A.v, false, Bg.v,
                                                   (short)0, Cg, false, false);
      Cu = __builtin_amdgcn_wmma_f32_16x16x32_bf16(false, A.v, false, Bu.v,
                                                   (short)0, Cu, false, false);
      __syncthreads();   // cur fully read + next fully written
    }
    // fused SiLU(g)*u epilogue -> sH (bf16). C layout: VGPR r -> M=r / 8+r.
    // Hardware v_rcp_f32 instead of IEEE div: result is rounded to bf16.
    {
      int ncol  = nb + n_sub * 16 + l15;
      int rbase = m_sub * 16 + ((lane & 16) ? 8 : 0);
      #pragma unroll
      for (int r = 0; r < 8; ++r) {
        float g = Cg[r];
        float u = Cu[r];
        float s = __builtin_amdgcn_rcpf(1.f + __expf(-g));
        sH[rbase + r][ncol] = f2bf(g * s * u);
      }
    }
  }
  __syncthreads();

  // ================= Phase B: OUT = H * Wd, scatter to token rows ==========
  for (int nb = 0; nb < HID; nb += N_TILE) {
    floatx8 C = {0.f, 0.f, 0.f, 0.f, 0.f, 0.f, 0.f, 0.f};
    stageD(0, 0, nb);
    __syncthreads();
    int buf = 0;
    for (int kb = 0; kb < IE; kb += K_TILE, buf ^= 1) {
      if (kb + K_TILE < IE) stageD(buf ^ 1, kb + K_TILE, nb);    // prefetch

      BF16x16 A, Bd;
      const unsigned short* ap = &sH[m_sub * 16 + l15][kb + khalf];
      A.u.lo = *(const us8_t*)(ap);
      A.u.hi = *(const us8_t*)(ap + 16);
      const unsigned short* bp = &sB[buf][0][n_sub * 16 + l15][khalf];
      Bd.u.lo = *(const us8_t*)(bp);
      Bd.u.hi = *(const us8_t*)(bp + 16);

      C = __builtin_amdgcn_wmma_f32_16x16x32_bf16(false, A.v, false, Bd.v,
                                                  (short)0, C, false, false);
      __syncthreads();
    }
    {
      int ncol  = nb + n_sub * 16 + l15;
      int rbase = m_sub * 16 + ((lane & 16) ? 8 : 0);
      #pragma unroll
      for (int r = 0; r < 8; ++r) {
        int tok = sIdx[rbase + r];
        if (tok >= 0) out[(size_t)tok * HID + ncol] = C[r];
      }
    }
  }
}

// ---------------------------------------------------------------------------
extern "C" void kernel_launch(void* const* d_in, const int* in_sizes, int n_in,
                              void* d_out, int out_size, void* d_ws, size_t ws_size,
                              hipStream_t stream) {
  const float* x    = (const float*)d_in[0];   // [4,4096,2048] f32
  const int*   tok  = (const int*)d_in[1];     // [4,4096] int
  const float* gate = (const float*)d_in[2];   // [8,2048,1024] f32
  const float* up   = (const float*)d_in[3];   // [8,2048,1024] f32
  const float* down = (const float*)d_in[4];   // [8,1024,2048] f32
  float* out = (float*)d_out;                  // [4,4096,2048] f32

  int* cnt    = (int*)d_ws;                    // [8] counters
  int* bucket = cnt + 16;                      // [8][16384] token indices

  zero_counts_kernel<<<1, 32, 0, stream>>>(cnt);
  route_kernel<<<(T_TOK + 255) / 256, 256, 0, stream>>>(tok, cnt, bucket);

  dim3 grid(T_TOK / M_TILE, NUM_EXPERTS);      // empty tiles exit immediately
  moe_mlp_kernel<<<grid, THREADS, 0, stream>>>(x, gate, up, down, cnt, bucket, out);
}
